// MultiHeadedAttention_1065151889522
// MI455X (gfx1250) — matrix operand order repaired and checked by reading, not compiled, not run on hardware
//
#include <hip/hip_runtime.h>

typedef _Float16 half_t;
typedef __attribute__((ext_vector_type(16))) _Float16 v16h;
typedef __attribute__((ext_vector_type(8)))  _Float16 v8h;
typedef __attribute__((ext_vector_type(8)))  float    v8f;
typedef __attribute__((ext_vector_type(4)))  float    v4f;

#define D_MODEL 1024
#define SEQ     2048
#define BATCH   2
#define HEADS   16
#define DEPTH   64

static __device__ __forceinline__ v8f wmma16(v16h a, v16h b, v8f c) {
  // D = A(16x32 f16) * B(32x16 f16) + C(16x16 f32)
  return __builtin_amdgcn_wmma_f32_16x16x32_f16(false, a, false, b, (short)0, c,
                                                false, false);
}
static __device__ __forceinline__ v16h cat16(v8h a, v8h b) {
  return __builtin_shufflevector(a, b, 0,1,2,3,4,5,6,7,8,9,10,11,12,13,14,15);
}

// ---------------------------------------------------------------------------
// Projection: out = x @ W^T (f32 in, f16 out), LDS-staged.
//   Block 256 thr = 8 waves as 4(M) x 2(N); block tile 64x64, K-step 32.
//   Staging: 256 threads load 64x32 f32 of A and of B, convert to f16 in LDS.
//   Each wave computes two 16x16 tiles (same A fragment, two B fragments).
//   VLAYOUT==0: out[((b*16+head)*2048 + s)*64 + d]   (Q,K head-split)
//   VLAYOUT==1: out[((b*16+head)*64 + d)*2048 + s]   (V transposed)
// ---------------------------------------------------------------------------
template <int VLAYOUT>
__global__ void mha_proj_f16_kernel(const float* __restrict__ x,
                                    const float* __restrict__ w,
                                    half_t* __restrict__ out) {
  __shared__ alignas(16) half_t Ash[64 * 32];
  __shared__ alignas(16) half_t Bsh[64 * 32];

  const int tid  = threadIdx.x;
  const int lane = tid & 31;
  const int wv   = tid >> 5;           // 0..7
  const int wm   = wv >> 1;            // 0..3  (M tile row)
  const int wn   = wv & 1;             // 0..1  (N 32-col group)
  const int m    = lane & 15;
  const int hh   = lane >> 4;

  const int row0 = blockIdx.x * 64;
  const int col0 = blockIdx.y * 64;

  // staging map: thread -> (row 0..63, 8-float k-chunk)
  const int srow = tid >> 2;           // 0..63
  const int scg  = (tid & 3) * 8;      // 0,8,16,24
  const float* ap = x + (size_t)(row0 + srow) * D_MODEL + scg;
  const float* bp = w + (size_t)(col0 + srow) * D_MODEL + scg;

  v8f acc0 = {}, acc1 = {};

  for (int kk = 0; kk < D_MODEL; kk += 32) {
    __syncthreads();                   // protect LDS from previous-iter readers
    __builtin_prefetch(ap + kk + 32, 0, 1);
    __builtin_prefetch(bp + kk + 32, 0, 1);
    v4f a0 = *(const v4f*)(ap + kk);
    v4f a1 = *(const v4f*)(ap + kk + 4);
    v4f b0 = *(const v4f*)(bp + kk);
    v4f b1 = *(const v4f*)(bp + kk + 4);
    v8h ah, bh8;
#pragma unroll
    for (int i = 0; i < 4; ++i) {
      ah[i]      = (half_t)a0[i];
      ah[i + 4]  = (half_t)a1[i];
      bh8[i]     = (half_t)b0[i];
      bh8[i + 4] = (half_t)b1[i];
    }
    *(v8h*)(Ash + srow * 32 + scg) = ah;
    *(v8h*)(Bsh + srow * 32 + scg) = bh8;
    __syncthreads();

    // A fragment: rows (wm*16 + m), contiguous K = 16h..16h+15
    const half_t* arow = Ash + (wm * 16 + m) * 32;
    const v16h A = cat16(*(const v8h*)(arow + 16 * hh),
                         *(const v8h*)(arow + 16 * hh + 8));
    // B fragments: cols (wn*32 + t*16 + m), contiguous K = 16h..16h+15
    const half_t* br0 = Bsh + (wn * 32 + m) * 32;
    const half_t* br1 = Bsh + (wn * 32 + 16 + m) * 32;
    const v16h B0 = cat16(*(const v8h*)(br0 + 16 * hh),
                          *(const v8h*)(br0 + 16 * hh + 8));
    const v16h B1 = cat16(*(const v8h*)(br1 + 16 * hh),
                          *(const v8h*)(br1 + 16 * hh + 8));
    acc0 = wmma16(A, B0, acc0);
    acc1 = wmma16(A, B1, acc1);
  }

  // store: C layout VGPR r -> M = r + 8*hh, N = m
#pragma unroll
  for (int t = 0; t < 2; ++t) {
    const v8f acc = t ? acc1 : acc0;
    const int gn   = col0 + wn * 32 + t * 16 + m;
    const int head = gn >> 6;
    const int d    = gn & 63;
#pragma unroll
    for (int r = 0; r < 8; ++r) {
      const int gm = row0 + wm * 16 + r + 8 * hh;
      const int b  = gm >> 11;
      const int s  = gm & 2047;
      size_t idx;
      if (VLAYOUT)
        idx = ((size_t)(b * HEADS + head) * DEPTH + d) * SEQ + s;
      else
        idx = ((size_t)(b * HEADS + head) * SEQ + s) * DEPTH + d;
      out[idx] = (half_t)acc[r];
    }
  }
}

// ---------------------------------------------------------------------------
// Attention: one block (256 thr = 8 waves) per (bh, 16-row q-block).
// LDS: sc f32[16][2048] | P f16[16][2048] | red f32[16][16] | rowsum f32[16]
// Phase 1 stores RAW scores (no scale -> no WMMA->VALU hazard nops);
// the 1/sqrt(64) scale is folded into the softmax exp as an FMA.
// Phase 3 reuses sc as the cross-wave partial-accumulator scratch.
// ---------------------------------------------------------------------------
#define SC_BYTES   (16 * SEQ * 4)            // 131072
#define P_BYTES    (16 * SEQ * 2)            // 65536
#define RED_OFF    (SC_BYTES + P_BYTES)      // 196608
#define RSUM_OFF   (RED_OFF + 1024)          // 197632
#define SH_BYTES   (RSUM_OFF + 64)           // 197696

__global__ void mha_attn_kernel(const half_t* __restrict__ Qh,
                                const half_t* __restrict__ Kh,
                                const half_t* __restrict__ Vt,
                                float* __restrict__ out) {
  extern __shared__ char smem[];
  float*  sc     = (float*)smem;
  half_t* P      = (half_t*)(smem + SC_BYTES);
  float*  red    = (float*)(smem + RED_OFF);
  float*  rowsum = (float*)(smem + RSUM_OFF);

  const int bh   = blockIdx.x;         // b*16 + head
  const int q0   = blockIdx.y * 16;
  const int tid  = threadIdx.x;        // 0..255
  const int lane = tid & 31;
  const int w    = tid >> 5;           // wave 0..7
  const int m    = lane & 15;
  const int hh   = lane >> 4;

  // --- Q fragments (A operand): d in [0,32) and [32,64) ---
  const half_t* qrow = Qh + ((size_t)bh * SEQ + q0 + m) * DEPTH;
  const v16h qa0 = cat16(*(const v8h*)(qrow + 16 * hh),
                         *(const v8h*)(qrow + 16 * hh + 8));
  const v16h qa1 = cat16(*(const v8h*)(qrow + 32 + 16 * hh),
                         *(const v8h*)(qrow + 32 + 16 * hh + 8));

  // --- Phase 1: raw scores = Q K^T -> LDS; 8 waves x 16 k-tiles ---
  for (int t = w * 16; t < w * 16 + 16; ++t) {
    const int kc = t * 16;
    const half_t* krow = Kh + ((size_t)bh * SEQ + kc + m) * DEPTH;
    __builtin_prefetch(krow + 16 * DEPTH, 0, 1);
    const v16h b0 = cat16(*(const v8h*)(krow + 16 * hh),
                          *(const v8h*)(krow + 16 * hh + 8));
    const v16h b1 = cat16(*(const v8h*)(krow + 32 + 16 * hh),
                          *(const v8h*)(krow + 32 + 16 * hh + 8));
    v8f c = {};
    c = wmma16(qa0, b0, c);
    c = wmma16(qa1, b1, c);
#pragma unroll
    for (int r = 0; r < 8; ++r)
      sc[(r + 8 * hh) * SEQ + kc + m] = c[r];    // raw; scale folded into exp
  }
  __syncthreads();

  // --- Phase 2: softmax (16 rows x 2048), 16 lanes/row, 128 elems each ---
  {
    const int row = tid & 15;
    const int seg = tid >> 4;                    // 0..15
    float* srow = sc + row * SEQ + seg * 128;
    float mx = -3.4e38f;
    for (int i = 0; i < 128; ++i) mx = fmaxf(mx, srow[i]);
    red[row * 16 + seg] = mx;
    __syncthreads();
    float rm = red[row * 16];
#pragma unroll
    for (int i = 1; i < 16; ++i) rm = fmaxf(rm, red[row * 16 + i]);
    __syncthreads();                             // before red is overwritten

    const float rms = rm * 0.125f;               // scaled row max
    float sum = 0.f;
    half_t* prow = P + row * SEQ + seg * 128;
    for (int i = 0; i < 128; ++i) {
      const float e = __expf(__builtin_fmaf(srow[i], 0.125f, -rms));
      prow[i] = (half_t)e;                       // unnormalized exp
      sum += e;
    }
    red[row * 16 + seg] = sum;
    __syncthreads();
    if (tid < 16) {
      float s = 0.f;
#pragma unroll
      for (int i = 0; i < 16; ++i) s += red[tid * 16 + i];
      rowsum[tid] = s;
    }
    __syncthreads();                             // also: everyone done with sc
  }

  // --- Phase 3: out = P @ V; wave = (d-slice 0..3, k-half 0..1) ---
  const int slice = w & 3;
  const int khalf = w >> 2;
  const int dcol  = slice * 16 + m;
  const int k0    = khalf * (SEQ / 2);
  const half_t* vrow = Vt + ((size_t)bh * DEPTH + dcol) * SEQ;

  v8f acc = {};
  for (int kk = k0; kk < k0 + SEQ / 2; kk += 32) {
    const half_t* pr = P + m * SEQ + kk;
    const v16h A  = cat16(*(const v8h*)(pr + 16 * hh),
                          *(const v8h*)(pr + 16 * hh + 8));
    const v16h Bv = cat16(*(const v8h*)(vrow + kk + 16 * hh),
                          *(const v8h*)(vrow + kk + 16 * hh + 8));
    acc = wmma16(A, Bv, acc);
  }

  // cross-wave reduction of the two k-halves; sc is dead, reuse it
  float* part = sc;                              // [4 slices][16 M][16 N]
  if (khalf == 1) {
#pragma unroll
    for (int r = 0; r < 8; ++r)
      part[slice * 256 + (r + 8 * hh) * 16 + m] = acc[r];
  }
  __syncthreads();

  if (khalf == 0) {
    const int b    = bh >> 4;
    const int head = bh & 15;
#pragma unroll
    for (int r = 0; r < 8; ++r) {
      const int M = r + 8 * hh;
      const float val =
          (acc[r] + part[slice * 256 + M * 16 + m]) * (1.0f / rowsum[M]);
      out[(size_t)(b * SEQ + q0 + M) * D_MODEL + head * DEPTH + dcol] = val;
    }
  }
}

// ---------------------------------------------------------------------------
extern "C" void kernel_launch(void* const* d_in, const int* in_sizes, int n_in,
                              void* d_out, int out_size, void* d_ws, size_t ws_size,
                              hipStream_t stream) {
  (void)in_sizes; (void)n_in; (void)out_size; (void)ws_size;
  const float* q  = (const float*)d_in[0];
  const float* k  = (const float*)d_in[1];
  const float* v  = (const float*)d_in[2];
  const float* wq = (const float*)d_in[3];
  const float* wk = (const float*)d_in[4];
  const float* wv = (const float*)d_in[5];
  float* out = (float*)d_out;

  const size_t perT = (size_t)BATCH * HEADS * SEQ * DEPTH;   // 4M halfs = 8MB
  half_t* Qh = (half_t*)d_ws;
  half_t* Kh = Qh + perT;
  half_t* Vt = Kh + perT;

  // Projections: grid 64 x 16, block 256 (64x64 tile, LDS-staged)
  dim3 pg(64, 16, 1), pb(256, 1, 1);
  mha_proj_f16_kernel<0><<<pg, pb, 0, stream>>>(q, wq, Qh);
  mha_proj_f16_kernel<0><<<pg, pb, 0, stream>>>(k, wk, Kh);
  mha_proj_f16_kernel<1><<<pg, pb, 0, stream>>>(v, wv, Vt);

  // Attention: grid (bh=32, qblocks=128), block 256, ~197KB dynamic LDS
  (void)hipFuncSetAttribute(reinterpret_cast<const void*>(&mha_attn_kernel),
                            hipFuncAttributeMaxDynamicSharedMemorySize,
                            (int)SH_BYTES);
  mha_attn_kernel<<<dim3(BATCH * HEADS, SEQ / 16, 1), dim3(256, 1, 1),
                    SH_BYTES, stream>>>(Qh, Kh, Vt, out);
}
